// NegativeContrastiveLoss_2508260901226
// MI455X (gfx1250) — matrix operand order repaired and checked by reading, not compiled, not run on hardware
//
#include <hip/hip_runtime.h>
#include <math.h>

#define K_DIM 256
#define TAU_INV 10.0f
#define EPS 1e-8f

typedef __attribute__((ext_vector_type(16))) _Float16 v16h;
typedef __attribute__((ext_vector_type(8)))  _Float16 v8h;
typedef __attribute__((ext_vector_type(4)))  _Float16 v4h;
typedef __attribute__((ext_vector_type(8)))  float    v8f;
typedef __attribute__((ext_vector_type(4)))  float    v4f;

// ---------------------------------------------------------------------------
// Kernel 1: L2-normalize each row, emit fp16 copy for WMMA consumption.
// 1 wave per row: 32 lanes x 8 elements (two float4 loads per lane).
// ---------------------------------------------------------------------------
__global__ __launch_bounds__(256)
void normalize_rows(const float* __restrict__ x, _Float16* __restrict__ xh, int n) {
    const int wave = threadIdx.x >> 5;
    const int lane = threadIdx.x & 31;
    const int row  = blockIdx.x * 8 + wave;
    if (row >= n) return;

    const float* xr = x + (size_t)row * K_DIM;
    v4f a = *(const v4f*)(xr + lane * 4);
    v4f b = *(const v4f*)(xr + 128 + lane * 4);

    float ss = a[0]*a[0] + a[1]*a[1] + a[2]*a[2] + a[3]*a[3]
             + b[0]*b[0] + b[1]*b[1] + b[2]*b[2] + b[3]*b[3];
    #pragma unroll
    for (int m = 1; m < 32; m <<= 1) ss += __shfl_xor(ss, m, 32);

    const float scale = 1.0f / fmaxf(sqrtf(ss), 1e-12f);

    _Float16* out = xh + (size_t)row * K_DIM;
    v4h h0, h1;
    #pragma unroll
    for (int i = 0; i < 4; ++i) { h0[i] = (_Float16)(a[i] * scale); h1[i] = (_Float16)(b[i] * scale); }
    *(v4h*)(out + lane * 4)       = h0;
    *(v4h*)(out + 128 + lane * 4) = h1;
}

// ---------------------------------------------------------------------------
// Kernel 2: fused tile-GEMM (xn @ xn^T) + exp + diagonal-skip + row-sum.
// One wave per (16-row tile, 512-col chunk). A frags held in registers for
// the full K=256; B frags streamed per 16-col tile. Epilogue accumulates
// exp(10*dot) per lane across 32 column tiles, then half-wave reduce +
// one atomicAdd per row.
//
// VGPR layouts per CDNA5 ISA 7.12.2 (wave32):
//   A 16x32 f16 : lane = m + 16*sel ; elem i -> k = sel*8 + i (+8 if i>=8)
//   B 32x16 f16 : lane = n + 16*sel ; elem i -> k = sel*8 + i (+8 if i>=8)
//   C 16x16 f32 : vgpr r, lane = n + 16*hi -> M = r + 8*hi, N = n
// ---------------------------------------------------------------------------
__global__ __launch_bounds__(32)
void gemm_rowsum(const _Float16* __restrict__ xh, float* __restrict__ rowsum, int n) {
    const int lane = threadIdx.x & 31;
    const int m16  = lane & 15;
    const int sel  = lane >> 4;
    const int rowbase  = blockIdx.x * 16;
    const int colbase0 = blockIdx.y * 512;

    union Frag { v16h v; v8h h[2]; };

    // Preload all 8 A fragments (K blocks of 32) for this 16-row strip.
    Frag A[8];
    const _Float16* arow = xh + (size_t)(rowbase + m16) * K_DIM;
    #pragma unroll
    for (int kb = 0; kb < 8; ++kb) {
        A[kb].h[0] = *(const v8h*)(arow + kb * 32 + sel * 8);        // k = sel*8 + 0..7
        A[kb].h[1] = *(const v8h*)(arow + kb * 32 + 16 + sel * 8);   // k = 16 + sel*8 + 0..7
    }

    float acc[8];
    #pragma unroll
    for (int r = 0; r < 8; ++r) acc[r] = 0.0f;

    for (int ct = 0; ct < 32; ++ct) {
        const int colbase = colbase0 + ct * 16;
        const _Float16* brow = xh + (size_t)(colbase + m16) * K_DIM; // column n = m16 of B
        v8f c = {};
        #pragma unroll
        for (int kb = 0; kb < 8; ++kb) {
            Frag B;
            B.h[0] = *(const v8h*)(brow + kb * 32 + sel * 8);        // k = sel*8 + 0..7
            B.h[1] = *(const v8h*)(brow + kb * 32 + 16 + sel * 8);   // k = 16 + sel*8 + 0..7
            c = __builtin_amdgcn_wmma_f32_16x16x32_f16(
                    /*neg_a=*/false, A[kb].v, /*neg_b=*/false, B.v,
                    /*c_mod=*/(short)0, c, /*reuse_a=*/false, /*reuse_b=*/false);
        }
        // Epilogue: exp(cos/tau), skipping the diagonal element.
        #pragma unroll
        for (int r = 0; r < 8; ++r) {
            const int grow = rowbase + r + 8 * sel;
            const int gcol = colbase + m16;
            const float e = __expf(c[r] * TAU_INV);
            acc[r] += (grow == gcol) ? 0.0f : e;
        }
    }

    // Reduce across the 16 lanes of each half-wave (same row set per half).
    #pragma unroll
    for (int r = 0; r < 8; ++r) {
        float s = acc[r];
        s += __shfl_xor(s, 1, 32);
        s += __shfl_xor(s, 2, 32);
        s += __shfl_xor(s, 4, 32);
        s += __shfl_xor(s, 8, 32);
        if (m16 == 0) atomicAdd(&rowsum[rowbase + r + 8 * sel], s);
    }
    (void)n;
}

// ---------------------------------------------------------------------------
// Kernel 3: loss_i = log(rowsum_i/(n-1) + eps); out = mean(loss). One block.
// ---------------------------------------------------------------------------
__global__ __launch_bounds__(256)
void finalize_loss(const float* __restrict__ rowsum, float* __restrict__ out, int n) {
    __shared__ float sdata[256];
    const float invnm1 = 1.0f / (float)(n - 1);
    float local = 0.0f;
    for (int i = threadIdx.x; i < n; i += 256)
        local += logf(rowsum[i] * invnm1 + EPS);
    sdata[threadIdx.x] = local;
    __syncthreads();
    #pragma unroll
    for (int s = 128; s > 0; s >>= 1) {
        if (threadIdx.x < (unsigned)s) sdata[threadIdx.x] += sdata[threadIdx.x + s];
        __syncthreads();
    }
    if (threadIdx.x == 0) out[0] = sdata[0] / (float)n;
}

// ---------------------------------------------------------------------------
extern "C" void kernel_launch(void* const* d_in, const int* in_sizes, int n_in,
                              void* d_out, int out_size, void* d_ws, size_t ws_size,
                              hipStream_t stream) {
    const float* x = (const float*)d_in[0];
    const int n = in_sizes[0] / K_DIM;      // 8192

    // Workspace layout: [ fp16 normalized copy | fp32 row sums ]
    _Float16* xh     = (_Float16*)d_ws;
    float*    rowsum = (float*)((char*)d_ws + (size_t)n * K_DIM * sizeof(_Float16));

    hipMemsetAsync(rowsum, 0, (size_t)n * sizeof(float), stream);

    normalize_rows<<<(n + 7) / 8, 256, 0, stream>>>(x, xh, n);

    dim3 grid(n / 16, 16);                   // 512 row tiles x 16 column chunks
    gemm_rowsum<<<grid, 32, 0, stream>>>(xh, rowsum, n);

    finalize_loss<<<1, 256, 0, stream>>>(rowsum, (float*)d_out, n);
}
